// RelationalRGCN_86303072846108
// MI455X (gfx1250) — matrix operand order repaired and checked by reading, not compiled, not run on hardware
//
#include <hip/hip_runtime.h>
#include <hip/hip_fp16.h>

// Problem constants (match reference)
#define N_NODES 20000
#define N_EDGES 320000
#define N_REL   8
#define N_GRAPH 2
#define SLOPE   0.2f

typedef __attribute__((ext_vector_type(16))) _Float16 v16h;
typedef __attribute__((ext_vector_type(8)))  float    v8f;

union FragH { v16h v; uint4 u[2]; };

// ---------------------------------------------------------------------------
// Conversion kernels (fp32 -> fp16, weight transposes). Tiny, run once/layer.
// ---------------------------------------------------------------------------
__global__ void f32_to_f16_kernel(const float* __restrict__ in,
                                  _Float16* __restrict__ out, int n) {
  int i = blockIdx.x * blockDim.x + threadIdx.x;
  if (i < n) out[i] = (_Float16)in[i];
}

// Wt[n, r*cin + c] = W[r, c, n]   (W is (R, Cin, Cout) row-major), K1 = 1<<k1Sh
__global__ void convert_w_kernel(const float* __restrict__ W,
                                 _Float16* __restrict__ Wt, int k1Sh, int cout,
                                 int total) {
  int i = blockIdx.x * blockDim.x + threadIdx.x;
  if (i >= total) return;
  int n  = i >> k1Sh;
  int kc = i & ((1 << k1Sh) - 1);      // kc = r*cin + c
  Wt[i] = (_Float16)W[(size_t)kc * cout + n];
}

// Rt[n, c] = root[c, n]   (root is (Cin, Cout) row-major), cin = 1<<cinSh
__global__ void convert_root_kernel(const float* __restrict__ root,
                                    _Float16* __restrict__ Rt, int cinSh, int cout,
                                    int total) {
  int i = blockIdx.x * blockDim.x + threadIdx.x;
  if (i >= total) return;
  int n = i >> cinSh;
  int c = i & ((1 << cinSh) - 1);
  Rt[i] = (_Float16)root[(size_t)c * cout + n];
}

// ---------------------------------------------------------------------------
// Edge scatter: agg[dst, etype, :] += h_f16[src, :]   (packed f16 atomics)
// One block per edge; edge indices are block-uniform -> scalar loads.
// blockDim.x = cin/2 (64 or 128)
// ---------------------------------------------------------------------------
__global__ void scatter_kernel(const unsigned int* __restrict__ hsrc2,
                               const int* __restrict__ src,
                               const int* __restrict__ dst,
                               const int* __restrict__ et,
                               __half2* __restrict__ agg2,
                               float* __restrict__ cnt, int c2) {
  int e = blockIdx.x;
  int s = src[e];
  int d = dst[e];
  int r = et[e];
  unsigned int bits = hsrc2[(size_t)s * c2 + threadIdx.x];
  __half2 v = __builtin_bit_cast(__half2, bits);
  atomicAdd(agg2 + (size_t)(d * N_REL + r) * c2 + threadIdx.x, v);   // global_atomic_pk_add_f16
  if (threadIdx.x == 0) atomicAdd(cnt + d * N_REL + r, 1.0f);
}

// ---------------------------------------------------------------------------
// Segment mean: agg[n,r,:] /= max(cnt[n,r], 1)     (cin = 1<<cinSh)
// ---------------------------------------------------------------------------
__global__ void mean_kernel(_Float16* __restrict__ agg,
                            const float* __restrict__ cnt, int cinSh, int total) {
  int i = blockIdx.x * blockDim.x + threadIdx.x;
  if (i >= total) return;
  float c   = cnt[i >> cinSh];
  float inv = 1.0f / fmaxf(c, 1.0f);
  agg[i] = (_Float16)((float)agg[i] * inv);
}

// ---------------------------------------------------------------------------
// WMMA GEMM: out(N, cout) = agg(N, K1) @ Wt^T + H(N, cin) @ Rt^T + bias, LeakyReLU
// Wt is (cout, K1) row-major; Rt is (cout, cin) row-major (pre-transposed B).
// Each wave: 16x64 tile (4 accumulators). 8 waves/block -> 128x64 block tile.
// k-loop unrolled x2 with named ping-pong buffers: loads for the next 32-wide
// k-step are issued before each WMMA group, and no register copies are needed
// at the swap point.
// ---------------------------------------------------------------------------
__device__ __forceinline__ v16h load_a_at(const _Float16* __restrict__ pA, int k0) {
  // pA already points at row m, half-group K offset applied; elements
  // 0..7 = K[k0..k0+7], 8..15 = K[k0+16..k0+23] (16-bit A layout, M-major).
  FragH f;
  f.u[0] = *(const uint4*)(pA + k0);
  f.u[1] = *(const uint4*)(pA + k0 + 16);
  return f.v;
}

__device__ __forceinline__ v16h load_b_at(const _Float16* __restrict__ pB, int k0) {
  // pB points at column n's row in Bt with half-group K offset applied;
  // elements 0..7 = K[k0..k0+7], 8..15 = K[k0+16..k0+23] (same as A).
  FragH f;
  f.u[0] = *(const uint4*)(pB + k0);
  f.u[1] = *(const uint4*)(pB + k0 + 16);
  return f.v;
}

__global__ __launch_bounds__(256)
void rgcn_gemm_kernel(const _Float16* __restrict__ A,   // (N, K1) agg, f16
                      const _Float16* __restrict__ Wt,  // (cout, K1) f16
                      const _Float16* __restrict__ H,   // (N, cin) f16
                      const _Float16* __restrict__ Rt,  // (cout, cin) f16
                      const float* __restrict__ bias,   // (cout,)
                      float* __restrict__ out,          // (N, cout) f32
                      int nrows, int K1, int cin, int cout) {
  const int lane = threadIdx.x & 31;
  const int wave = threadIdx.x >> 5;
  const int m0     = blockIdx.x * 128 + wave * 16;
  const int n_base = blockIdx.y * 64;
  const int maxRow = nrows - 1;

  int m = m0 + (lane & 15);
  if (m > maxRow) m = maxRow;              // clamp loads; stores are guarded
  const int aOff = (lane >> 4) << 3;       // A half-group K offset (0 or 8)
  const int bOff = (lane >> 4) << 3;       // B half-group K offset (0 or 8)
  const int nCol = lane & 15;

  v8f acc[4] = {};

  // Pipelined GEMM pass: acc += A(N,K)[m0:,:] * Bt(cols,K)^T.  K % 64 == 0.
  auto gemm_pass = [&](const _Float16* __restrict__ Abase, int lda,
                       const _Float16* __restrict__ Bbase, int ldb, int K) {
    const _Float16* pA  = Abase + (size_t)m * lda + aOff;
    const _Float16* pB0 = Bbase + (size_t)(n_base +  0 + nCol) * ldb + bOff;
    const _Float16* pB1 = Bbase + (size_t)(n_base + 16 + nCol) * ldb + bOff;
    const _Float16* pB2 = Bbase + (size_t)(n_base + 32 + nCol) * ldb + bOff;
    const _Float16* pB3 = Bbase + (size_t)(n_base + 48 + nCol) * ldb + bOff;

    // buffer 0 primed at k = 0
    v16h a0  = load_a_at(pA, 0);
    v16h b00 = load_b_at(pB0, 0);
    v16h b10 = load_b_at(pB1, 0);
    v16h b20 = load_b_at(pB2, 0);
    v16h b30 = load_b_at(pB3, 0);

    for (int k0 = 0; k0 < K; k0 += 64) {
      // stage 1: issue loads for k0+32 into buffer 1, compute buffer 0
      v16h a1  = load_a_at(pA, k0 + 32);
      v16h b01 = load_b_at(pB0, k0 + 32);
      v16h b11 = load_b_at(pB1, k0 + 32);
      v16h b21 = load_b_at(pB2, k0 + 32);
      v16h b31 = load_b_at(pB3, k0 + 32);
      __builtin_prefetch(pA + k0 + 64, 0, 1);     // global_prefetch_b8
      acc[0] = __builtin_amdgcn_wmma_f32_16x16x32_f16(false, a0, false, b00,
                                                      (short)0, acc[0], false, false);
      acc[1] = __builtin_amdgcn_wmma_f32_16x16x32_f16(false, a0, false, b10,
                                                      (short)0, acc[1], true, false);
      acc[2] = __builtin_amdgcn_wmma_f32_16x16x32_f16(false, a0, false, b20,
                                                      (short)0, acc[2], true, false);
      acc[3] = __builtin_amdgcn_wmma_f32_16x16x32_f16(false, a0, false, b30,
                                                      (short)0, acc[3], true, false);

      // stage 2: issue loads for k0+64 into buffer 0 (tail-guarded), compute buffer 1
      if (k0 + 64 < K) {
        a0  = load_a_at(pA, k0 + 64);
        b00 = load_b_at(pB0, k0 + 64);
        b10 = load_b_at(pB1, k0 + 64);
        b20 = load_b_at(pB2, k0 + 64);
        b30 = load_b_at(pB3, k0 + 64);
      }
      acc[0] = __builtin_amdgcn_wmma_f32_16x16x32_f16(false, a1, false, b01,
                                                      (short)0, acc[0], false, false);
      acc[1] = __builtin_amdgcn_wmma_f32_16x16x32_f16(false, a1, false, b11,
                                                      (short)0, acc[1], true, false);
      acc[2] = __builtin_amdgcn_wmma_f32_16x16x32_f16(false, a1, false, b21,
                                                      (short)0, acc[2], true, false);
      acc[3] = __builtin_amdgcn_wmma_f32_16x16x32_f16(false, a1, false, b31,
                                                      (short)0, acc[3], true, false);
    }
  };

  gemm_pass(A, K1,  Wt, K1,  K1);   // agg @ Wcat   (K = R*Cin)
  gemm_pass(H, cin, Rt, cin, cin);  // h   @ root   (K = Cin)

  // epilogue: bias + LeakyReLU(0.2); C layout: VGPR d -> M = d + (lane>>4)*8
  const int rbase = m0 + ((lane >> 4) << 3);
  const int cbase = n_base + nCol;
#pragma unroll
  for (int j = 0; j < 4; ++j) {
    const int col = cbase + 16 * j;
    const float bv = bias[col];
#pragma unroll
    for (int d = 0; d < 8; ++d) {
      const int row = rbase + d;
      if (row < nrows) {
        float v = acc[j][d] + bv;
        out[(size_t)row * cout + col] = (v > 0.0f) ? v : SLOPE * v;
      }
    }
  }
}

// ---------------------------------------------------------------------------
// Host launcher
// ---------------------------------------------------------------------------
extern "C" void kernel_launch(void* const* d_in, const int* in_sizes, int n_in,
                              void* d_out, int out_size, void* d_ws, size_t ws_size,
                              hipStream_t stream) {
  (void)in_sizes; (void)n_in; (void)out_size;

  const float* x  = (const float*)d_in[0];           // (B, N, 128)
  const int*   ei = (const int*)d_in[1];             // (B, 2, E)
  const int*   et = (const int*)d_in[2];             // (B, E)
  const float* Wl[3]    = { (const float*)d_in[3], (const float*)d_in[6], (const float*)d_in[9]  };
  const float* rootl[3] = { (const float*)d_in[4], (const float*)d_in[7], (const float*)d_in[10] };
  const float* biasl[3] = { (const float*)d_in[5], (const float*)d_in[8], (const float*)d_in[11] };
  const int dims[4] = {128, 256, 256, 128};

  // Workspace carve-up (256B aligned slices)
  char*  base = (char*)d_ws;
  size_t off  = 0;
  auto alloc = [&](size_t bytes) -> void* {
    void* p = base + off;
    off += (bytes + 255) & ~(size_t)255;
    return p;
  };
  float*    hA    = (float*)   alloc((size_t)N_NODES * 256 * 4);
  float*    hB    = (float*)   alloc((size_t)N_NODES * 256 * 4);
  _Float16* aggh  = (_Float16*)alloc((size_t)N_NODES * N_REL * 256 * 2);
  float*    cnt   = (float*)   alloc((size_t)N_NODES * N_REL * 4);
  _Float16* hf16  = (_Float16*)alloc((size_t)N_NODES * 256 * 2);
  _Float16* wt16  = (_Float16*)alloc((size_t)N_REL * 256 * 256 * 2);
  _Float16* rt16  = (_Float16*)alloc((size_t)256 * 256 * 2);
  if (off > ws_size) return;  // workspace too small; nothing safe to do

  float* outF = (float*)d_out;

  for (int b = 0; b < N_GRAPH; ++b) {
    const int* srcp = ei + (size_t)b * 2 * N_EDGES;
    const int* dstp = srcp + N_EDGES;
    const int* etp  = et + (size_t)b * N_EDGES;

    for (int l = 0; l < 3; ++l) {
      const int cin  = dims[l];
      const int cout = dims[l + 1];
      const int K1   = N_REL * cin;
      const int cinSh = (cin == 128) ? 7 : 8;
      const int k1Sh  = cinSh + 3;   // K1 = 8 * cin

      const float* hin  = (l == 0) ? (x + (size_t)b * N_NODES * 128)
                                   : ((l == 1) ? hA : hB);
      float*       hout = (l == 2) ? (outF + (size_t)b * N_NODES * 128)
                                   : ((l == 0) ? hA : hB);

      // weights -> fp16, pre-transposed for contiguous B-fragment loads
      {
        int tot = cout * K1;
        convert_w_kernel<<<(tot + 255) / 256, 256, 0, stream>>>(Wl[l], wt16, k1Sh, cout, tot);
        tot = cout * cin;
        convert_root_kernel<<<(tot + 255) / 256, 256, 0, stream>>>(rootl[l], rt16, cinSh, cout, tot);
      }

      // zero aggregation buffers
      hipMemsetAsync(aggh, 0, (size_t)N_NODES * N_REL * cin * 2, stream);
      hipMemsetAsync(cnt,  0, (size_t)N_NODES * N_REL * 4, stream);

      // h -> fp16 messages
      {
        int tot = N_NODES * cin;
        f32_to_f16_kernel<<<(tot + 255) / 256, 256, 0, stream>>>(hin, hf16, tot);
      }

      // scatter-add per (dst, relation); packed f16 atomics
      scatter_kernel<<<N_EDGES, cin / 2, 0, stream>>>(
          (const unsigned int*)hf16, srcp, dstp, etp,
          (__half2*)aggh, cnt, cin / 2);

      // segment mean
      {
        int tot = N_NODES * N_REL * cin;
        mean_kernel<<<(tot + 255) / 256, 256, 0, stream>>>(aggh, cnt, cinSh, tot);
      }

      // fused GEMM: agg@Wcat + h@root + bias, LeakyReLU
      dim3 grid((N_NODES + 127) / 128, cout / 64);
      rgcn_gemm_kernel<<<grid, 256, 0, stream>>>(
          aggh, wt16, hf16, rt16, biasl[l], hout, N_NODES, K1, cin, cout);
    }
  }
}
